// HyperImplicitField_86870008529436
// MI455X (gfx1250) — matrix operand order repaired and checked by reading, not compiled, not run on hardware
//
#include <hip/hip_runtime.h>
#include <hip/hip_bf16.h>
#include <math.h>

typedef _Float16 v16h __attribute__((ext_vector_type(16)));
typedef _Float16 v8h  __attribute__((ext_vector_type(8)));
typedef float    v8f  __attribute__((ext_vector_type(8)));

#define EPSF 1e-5f

// ---- layer geometry ----
// l=0: 63->256   Kpad=64  Npad=256
// l=1: 256->256  Kpad=256 Npad=256
// l=2: 256->256  Kpad=256 Npad=256
// l=3: 256->4    Kpad=256 Npad=16
__device__ __constant__ int d_POUT[4]  = {16384, 65792, 65792, 1028};
__device__ __constant__ int d_DIR[4]   = {63, 256, 256, 256};
__device__ __constant__ int d_DOR[4]   = {256, 256, 256, 4};
__device__ __constant__ int d_NTL[4]   = {16, 16, 16, 1};
__device__ __constant__ int d_KN[4]    = {16384, 65536, 65536, 4096};
__device__ __constant__ int d_WOFF[4]  = {0, 65536, 327680, 589824};
__device__ __constant__ int d_NPAD[4]  = {256, 256, 256, 16};
__device__ __constant__ int d_BOFF[4]  = {0, 1024, 2048, 3072};

#define WFRAG_HALVES 606208
#define BIAS_FLOATS  3136
#define HYPER_ROWS   595984   // 4 * (16384+65792+65792+1028)

// ---- workspace layout (bytes) ----
#define WS_COUNTS   0u
#define WS_CURSORS  16u
#define WS_OFFSETS  32u
#define WS_TILEOFF  64u
#define WS_H2       128u                       // 16*256 f32 = 16384
#define WS_BIAS     16512u                     // 3136 f32   = 12544
#define WS_WFRAG    29056u                     // 606208 f16 = 1212416
#define WS_PERM     1241472u                   // N ints

struct PTab {
    const float *W1[4], *b1[4], *g1[4], *be1[4];
    const float *W2[4], *b2[4], *g2[4], *be2[4];
    const float *W3[4], *b3[4];
};

__device__ __forceinline__ v8f wmma16x16x32(v16h a, v16h b, v8f c) {
    return __builtin_amdgcn_wmma_f32_16x16x32_f16(false, a, false, b, (short)0, c, false, false);
}
__device__ __forceinline__ v16h cat16(v8h lo, v8h hi) {
    return __builtin_shufflevector(lo, hi, 0,1,2,3,4,5,6,7,8,9,10,11,12,13,14,15);
}

// -------------------- utility kernels --------------------
__global__ void zero_ws_kernel(unsigned* a, int na, unsigned* b, int nb) {
    int g = blockIdx.x * blockDim.x + threadIdx.x;
    int stride = gridDim.x * blockDim.x;
    for (int j = g; j < na; j += stride) a[j] = 0u;
    for (int j = g; j < nb; j += stride) b[j] = 0u;
}

__global__ void hist_kernel(const int* __restrict__ idx, int n, int* counts) {
    int g = blockIdx.x * blockDim.x + threadIdx.x;
    if (g < n) atomicAdd(&counts[idx[g]], 1);
}

__global__ void scan_kernel(const int* __restrict__ counts, int* offsets, int* tileOff) {
    if (blockIdx.x == 0 && threadIdx.x == 0) {
        int o = 0, t = 0;
        offsets[0] = 0; tileOff[0] = 0;
        for (int b = 0; b < 4; b++) {
            o += counts[b];            offsets[b + 1] = o;
            t += (counts[b] + 31) / 32; tileOff[b + 1] = t;
        }
    }
}

__global__ void scatter_kernel(const int* __restrict__ idx, int n,
                               const int* __restrict__ offsets, int* cursors,
                               int* __restrict__ perm) {
    int g = blockIdx.x * blockDim.x + threadIdx.x;
    if (g < n) {
        int b = idx[g];
        int p = atomicAdd(&cursors[b], 1);
        perm[offsets[b] + p] = g;
    }
}

// -------------------- hypernet stage 1: c -> H2 (per layer,batch) --------------------
__global__ __launch_bounds__(256) void hyper_stage1(PTab tab, const float* __restrict__ c,
                                                    float* __restrict__ H2buf) {
    int lb = blockIdx.x, l = lb >> 2, b = lb & 3, t = threadIdx.x;
    __shared__ __align__(16) float cc[128];
    __shared__ __align__(16) float H[256];
    __shared__ float Ssum[256], Ssq[256];

    if (t < 128) cc[t] = c[b * 128 + t];
    __syncthreads();

    float h = tab.b1[l][t];
    {
        const float4* w = (const float4*)(tab.W1[l] + (size_t)t * 128);
        const float4* cv = (const float4*)cc;
        for (int j = 0; j < 32; j++) {
            float4 a = w[j], x = cv[j];
            h += a.x * x.x + a.y * x.y + a.z * x.z + a.w * x.w;
        }
    }
    Ssum[t] = h; Ssq[t] = h * h; __syncthreads();
    for (int s = 128; s > 0; s >>= 1) {
        if (t < s) { Ssum[t] += Ssum[t + s]; Ssq[t] += Ssq[t + s]; }
        __syncthreads();
    }
    float mu = Ssum[0] * (1.f / 256.f);
    float var = Ssq[0] * (1.f / 256.f) - mu * mu;
    float rstd = rsqrtf(var + EPSF);
    float hn = fmaxf((h - mu) * rstd * tab.g1[l][t] + tab.be1[l][t], 0.f);
    __syncthreads();
    H[t] = hn; __syncthreads();

    float h2 = tab.b2[l][t];
    {
        const float4* w = (const float4*)(tab.W2[l] + (size_t)t * 256);
        const float4* hv = (const float4*)H;
        for (int j = 0; j < 64; j++) {
            float4 a = w[j], x = hv[j];
            h2 += a.x * x.x + a.y * x.y + a.z * x.z + a.w * x.w;
        }
    }
    __syncthreads();
    Ssum[t] = h2; Ssq[t] = h2 * h2; __syncthreads();
    for (int s = 128; s > 0; s >>= 1) {
        if (t < s) { Ssum[t] += Ssum[t + s]; Ssq[t] += Ssq[t + s]; }
        __syncthreads();
    }
    mu = Ssum[0] * (1.f / 256.f);
    var = Ssq[0] * (1.f / 256.f) - mu * mu;
    rstd = rsqrtf(var + EPSF);
    float h2n = fmaxf((h2 - mu) * rstd * tab.g2[l][t] + tab.be2[l][t], 0.f);
    H2buf[lb * 256 + t] = h2n;
}

// -------------------- hypernet stage 2: H2 -> packed WMMA-B weights + biases --------------------
__global__ __launch_bounds__(256) void hyper_stage2(PTab tab, const float* __restrict__ H2buf,
                                                    _Float16* __restrict__ Wfrag,
                                                    float* __restrict__ biasWs) {
    int g = blockIdx.x * 256 + threadIdx.x;
    int l = -1, b = -1, r = 0, cum = 0;
    for (int L = 0; L < 4 && l < 0; L++) {
        for (int B = 0; B < 4; B++) {
            if (g < cum + d_POUT[L]) { l = L; b = B; r = g - cum; break; }
            cum += d_POUT[L];
        }
    }
    if (l < 0) return;

    const float4* H2 = (const float4*)(H2buf + (l * 4 + b) * 256);
    const float4* W3 = (const float4*)(tab.W3[l] + (size_t)r * 256);
    float acc = tab.b3[l][r];
    for (int j = 0; j < 64; j++) {
        float4 a = W3[j], x = H2[j];
        acc += a.x * x.x + a.y * x.y + a.z * x.z + a.w * x.w;
    }

    int di = d_DIR[l], doR = d_DOR[l];
    if (r < di * doR) {
        // B-matrix element (k=d, n=o) in WMMA B layout:
        // lane = (k%16 >= 8 ? 16 : 0) + n%16 ; half = (k%8) + (k%32 >= 16 ? 8 : 0) ; frag = kt*NT + nt
        int o = r / di, d = r % di;
        int kt = d >> 5, k32 = d & 31, nt = o >> 4;
        int lane = (((k32 >> 3) & 1) << 4) | (o & 15);
        size_t base = (size_t)d_WOFF[l] + (size_t)b * d_KN[l];
        size_t idx = base + ((size_t)(kt * d_NTL[l] + nt) * 32 + lane) * 16 + ((k32 & 7) | ((k32 >> 4) << 3));
        Wfrag[idx] = (_Float16)acc;
    } else {
        biasWs[d_BOFF[l] + b * d_NPAD[l] + (r - di * doR)] = acc;
    }
}

// -------------------- fused main MLP: 32-point tiles, WMMA f16 --------------------
#define SA 264   // hA row stride (halves): 528B rows keep 16B alignment, skew banks
#define SZ 260   // zS row stride (floats)

// One hidden layer: 32xK (f16, LDS) x Kx256 (f16, packed global) -> 32x256,
// +bias -> zS (f32 LDS) -> per-row LayerNorm+ReLU -> hA (f16 LDS).
// All geometry compile-time so nt-loop unrolls and acc[] stays in fixed VGPRs.
template<int K>
__device__ __forceinline__ void hidden_layer(const _Float16* __restrict__ WF,
                                             const float* __restrict__ BS,
                                             const _Float16* __restrict__ hArow,
                                             _Float16* hA, float* zS,
                                             int lane, int colQ, int kg,
                                             int rowBase, int tid) {
    v8f acc[4];
    #pragma unroll
    for (int i = 0; i < 4; i++) acc[i] = (v8f){0.f,0.f,0.f,0.f,0.f,0.f,0.f,0.f};

    const _Float16* wlane = WF + (size_t)lane * 16;
    #pragma unroll 2
    for (int kt = 0; kt < K / 32; kt++) {
        int koff = kt * 32 + kg * 8;
        v8h a0 = *(const v8h*)(hArow + koff);
        v8h a1 = *(const v8h*)(hArow + koff + 16);
        v16h a = cat16(a0, a1);
        #pragma unroll
        for (int i = 0; i < 4; i++) {
            int nt = colQ + 4 * i;
            const _Float16* bp = wlane + (size_t)(kt * 16 + nt) * 512;
            v16h bb = cat16(*(const v8h*)bp, *(const v8h*)(bp + 8));
            acc[i] = wmma16x16x32(a, bb, acc[i]);
        }
    }

    int mb = rowBase + kg * 8;   // C layout: lanes>=16 hold M = 8..15
    #pragma unroll
    for (int i = 0; i < 4; i++) {
        int ncol = (colQ + 4 * i) * 16 + (lane & 15);
        float bv = BS[ncol];
        #pragma unroll
        for (int j = 0; j < 8; j++)
            zS[(mb + j) * SZ + ncol] = acc[i][j] + bv;
    }
    __syncthreads();
    if (tid < 32) {
        float s = 0.f, s2 = 0.f;
        for (int c2 = 0; c2 < 256; c2++) {
            float v = zS[tid * SZ + c2];
            s += v; s2 += v * v;
        }
        float mu = s * (1.f / 256.f);
        float var = s2 * (1.f / 256.f) - mu * mu;
        float rstd = rsqrtf(var + EPSF);
        for (int c2 = 0; c2 < 256; c2++) {
            float v = (zS[tid * SZ + c2] - mu) * rstd;
            hA[tid * SA + c2] = (_Float16)fmaxf(v, 0.f);
        }
    }
    __syncthreads();
}

__global__ __launch_bounds__(256) void main_mlp(const float* __restrict__ x,
                                                const int* __restrict__ perm,
                                                const int* __restrict__ offsets,
                                                const int* __restrict__ tileOff,
                                                const _Float16* __restrict__ Wfrag,
                                                const float* __restrict__ biasWs,
                                                float* __restrict__ out) {
    __shared__ __align__(16) _Float16 hA[32 * SA];   // 16896 B
    __shared__ __align__(16) float    zS[32 * SZ];   // 33280 B
    __shared__ int rowN[32];
    __shared__ int meta[2];

    int tid = threadIdx.x;
    if (tid == 0) {
        int blk = blockIdx.x, b = -1, t = 0;
        if (blk < tileOff[4]) {
            for (int bb = 0; bb < 4; bb++)
                if (blk < tileOff[bb + 1]) { b = bb; t = blk - tileOff[bb]; break; }
        }
        meta[0] = b; meta[1] = t;
    }
    __syncthreads();
    int b = meta[0];
    if (b < 0) return;
    int t = meta[1];

    if (tid < 32) {
        int p = offsets[b] + t * 32 + tid;
        rowN[tid] = (p < offsets[b + 1]) ? perm[p] : -1;
    }
    __syncthreads();

    // ---- features: [x0,x1,x2, per-dim {sin(x_d*2^k*pi)}k0..9 then {cos}k0..9, 0-pad] ----
    {
        int row = tid >> 3, q = tid & 7;
        int n = rowN[row];
        float xv[3] = {0.f, 0.f, 0.f};
        if (n >= 0) { xv[0] = x[3 * n]; xv[1] = x[3 * n + 1]; xv[2] = x[3 * n + 2]; }
        const float PI_F = 3.14159265358979323846f;
        #pragma unroll
        for (int j = 0; j < 8; j++) {
            int col = q * 8 + j;
            float v;
            if (col < 3) v = xv[col];
            else if (col >= 63) v = 0.f;
            else {
                int e = col - 3, d = e / 20, rm = e % 20, k = rm % 10;
                float ang = xv[d] * ((float)(1 << k)) * PI_F;
                v = (rm < 10) ? sinf(ang) : cosf(ang);
            }
            hA[row * SA + col] = (_Float16)v;
        }
    }
    __syncthreads();

    int wave = tid >> 5, lane = tid & 31;
    int stripe = wave & 1, colQ = wave >> 1;     // 2 row-stripes x 4 col-quarters
    int rowBase = stripe * 16;
    int mrow = rowBase + (lane & 15);
    int kg = lane >> 4;
    const _Float16* hArow = hA + mrow * SA;

    // layers 0..2 (all compile-time geometry)
    hidden_layer< 64>(Wfrag +      0 + b * 16384, biasWs +    0 + b * 256,
                      hArow, hA, zS, lane, colQ, kg, rowBase, tid);
    hidden_layer<256>(Wfrag +  65536 + b * 65536, biasWs + 1024 + b * 256,
                      hArow, hA, zS, lane, colQ, kg, rowBase, tid);
    hidden_layer<256>(Wfrag + 327680 + b * 65536, biasWs + 2048 + b * 256,
                      hArow, hA, zS, lane, colQ, kg, rowBase, tid);

    // ---- final layer: K=256, one 16-col tile; owned by colQ==0 waves ----
    if (colQ == 0) {
        const _Float16* WF = Wfrag + 589824 + b * 4096;
        const float*    BS = biasWs + 3072 + b * 16;
        v8f acc = (v8f){0.f,0.f,0.f,0.f,0.f,0.f,0.f,0.f};
        const _Float16* wlane = WF + (size_t)lane * 16;
        #pragma unroll 2
        for (int kt = 0; kt < 8; kt++) {
            int koff = kt * 32 + kg * 8;
            v16h a = cat16(*(const v8h*)(hArow + koff),
                           *(const v8h*)(hArow + koff + 16));
            const _Float16* bp = wlane + (size_t)kt * 512;
            v16h bb = cat16(*(const v8h*)bp, *(const v8h*)(bp + 8));
            acc = wmma16x16x32(a, bb, acc);
        }
        int ncol = lane & 15;
        if (ncol < 4) {
            float bv = BS[ncol];
            int mb = rowBase + kg * 8;
            #pragma unroll
            for (int j = 0; j < 8; j++) {
                int n = rowN[mb + j];
                if (n >= 0) out[(size_t)n * 4 + ncol] = acc[j] + bv;
            }
        }
    }
}

// -------------------- host launcher --------------------
extern "C" void kernel_launch(void* const* d_in, const int* in_sizes, int n_in,
                              void* d_out, int out_size, void* d_ws, size_t ws_size,
                              hipStream_t stream) {
    const float* xp = (const float*)d_in[0];
    const int*   ip = (const int*)d_in[1];
    const float* cp = (const float*)d_in[2];
    int N = in_sizes[1];

    char* wsb = (char*)d_ws;
    if (ws_size < (size_t)WS_PERM + (size_t)N * 4) return;

    int*       counts  = (int*)(wsb + WS_COUNTS);
    int*       cursors = (int*)(wsb + WS_CURSORS);
    int*       offsets = (int*)(wsb + WS_OFFSETS);
    int*       tileOff = (int*)(wsb + WS_TILEOFF);
    float*     H2buf   = (float*)(wsb + WS_H2);
    float*     biasWs  = (float*)(wsb + WS_BIAS);
    _Float16*  Wfrag   = (_Float16*)(wsb + WS_WFRAG);
    int*       perm    = (int*)(wsb + WS_PERM);

    // params: 4 layers x 10 tensors at d_in[3..42]; detect flatten order by sizes
    // insertion: W1,b1,g1,be1,W2,b2,g2,be2,W3,b3   (in_sizes[4]==256)
    // sorted   : W1,W2,W3,b1,b2,b3,be1,be2,g1,g2   (in_sizes[4]==65536)
    bool ins = (n_in >= 5) && (in_sizes[4] == 256);
    PTab tab;
    for (int l = 0; l < 4; l++) {
        const float* const* L = (const float* const*)(d_in + 3 + l * 10);
        if (ins) {
            tab.W1[l]=L[0]; tab.b1[l]=L[1]; tab.g1[l]=L[2]; tab.be1[l]=L[3];
            tab.W2[l]=L[4]; tab.b2[l]=L[5]; tab.g2[l]=L[6]; tab.be2[l]=L[7];
            tab.W3[l]=L[8]; tab.b3[l]=L[9];
        } else {
            tab.W1[l]=L[0]; tab.W2[l]=L[1]; tab.W3[l]=L[2];
            tab.b1[l]=L[3]; tab.b2[l]=L[4]; tab.b3[l]=L[5];
            tab.be1[l]=L[6]; tab.be2[l]=L[7];
            tab.g1[l]=L[8]; tab.g2[l]=L[9];
        }
    }

    int nblk = (N + 255) / 256;
    zero_ws_kernel<<<256, 256, 0, stream>>>((unsigned*)(wsb + WS_COUNTS), 8,
                                            (unsigned*)(wsb + WS_BIAS),
                                            BIAS_FLOATS + WFRAG_HALVES / 2);
    hist_kernel<<<nblk, 256, 0, stream>>>(ip, N, counts);
    scan_kernel<<<1, 32, 0, stream>>>(counts, offsets, tileOff);
    scatter_kernel<<<nblk, 256, 0, stream>>>(ip, N, offsets, cursors, perm);
    hyper_stage1<<<16, 256, 0, stream>>>(tab, cp, H2buf);
    hyper_stage2<<<(HYPER_ROWS + 255) / 256, 256, 0, stream>>>(tab, H2buf, Wfrag, biasWs);
    main_mlp<<<(N + 31) / 32 + 4, 256, 0, stream>>>(xp, perm, offsets, tileOff,
                                                    Wfrag, biasWs, (float*)d_out);
    (void)out_size;
}
